// GPT_64269890617697
// MI455X (gfx1250) — matrix operand order repaired and checked by reading, not compiled, not run on hardware
//
#include <hip/hip_runtime.h>
#include <hip/hip_bf16.h>

// ---------------- model constants ----------------
#define B_ 2
#define T_ 1024
#define E_ 1024
#define H_ 16
#define L_ 12
#define D_ 64
#define V_ 50257
#define BT_ (B_ * T_)
#define E3_ (3 * E_)
#define E4_ (4 * E_)

typedef _Float16 half_t;
typedef __attribute__((ext_vector_type(16))) _Float16 v16h;
typedef __attribute__((ext_vector_type(8)))  _Float16 v8h;
typedef __attribute__((ext_vector_type(8)))  float    v8f;

#define KPAD 40   // 32 K-halves + 8 pad -> 80B row stride, conflict-free on 64 banks

// =====================================================================
// WMMA GEMM:  C[M][N] = A[M][K] (f16) * Bt[N][K]^T (f16)  + bias[N]
// Block tile 128x128, 256 threads = 8 waves (2x4), wave tile 64x32.
// Double-buffered LDS fed by GLOBAL_LOAD_ASYNC_TO_LDS_B128 (ASYNCcnt),
// one barrier per K-step, branchless loads (B row index clamped; the
// epilogue guards stores so garbage columns are never written).
// M must be a multiple of 128; K multiple of 32.
// =====================================================================
__global__ __launch_bounds__(256)
void gemm_bt_wmma(const half_t* __restrict__ A, const half_t* __restrict__ Bt,
                  float* __restrict__ C, const float* __restrict__ bias,
                  int M, int N, int K,
                  long long strideA, long long strideB, long long strideC,
                  int causal)
{
    const int m0 = blockIdx.y * 128;
    const int n0 = blockIdx.x * 128;
    if (causal && n0 > m0 + 127) return;   // fully masked block of causal scores

    A  += (long long)blockIdx.z * strideA;
    Bt += (long long)blockIdx.z * strideB;
    C  += (long long)blockIdx.z * strideC;

    __shared__ __align__(64) half_t As[2][128][KPAD];
    __shared__ __align__(64) half_t Bs[2][128][KPAD];

    const int tid  = threadIdx.x;
    const int lane = tid & 31;
    const int wave = tid >> 5;
    const int wm   = wave >> 2;   // 0..1  (64 rows each)
    const int wn   = wave & 3;    // 0..3  (32 cols each)

    v8f acc[4][2];
    #pragma unroll
    for (int i = 0; i < 4; ++i)
        #pragma unroll
        for (int j = 0; j < 2; ++j) acc[i][j] = {};

    // loaders: 256 threads, each owns 16 halves (32B) of the A and B tiles
    const int lrow = tid >> 1;         // 0..127
    const int lk   = (tid & 1) * 16;   // 0 or 16

    const int gn  = n0 + lrow;
    const int gnc = gn < N ? gn : N - 1;                 // clamped, branchless
    const half_t* aptr = A  + (size_t)(m0 + lrow) * K + lk;
    const half_t* bptr = Bt + (size_t)gnc * K + lk;

    const int nK = K >> 5;

    auto issue_tile = [&](int kt, int buf) {
        const unsigned av = (unsigned)(uintptr_t)&As[buf][lrow][lk];
        const unsigned bv = (unsigned)(uintptr_t)&Bs[buf][lrow][lk];
        const unsigned long long ga = (unsigned long long)(uintptr_t)(aptr + ((size_t)kt << 5));
        const unsigned long long gb = (unsigned long long)(uintptr_t)(bptr + ((size_t)kt << 5));
        // INST_OFFSET is added to BOTH the LDS and global address, so the
        // second 16B chunk of each 32B span is just offset:16.
        asm volatile(
            "global_load_async_to_lds_b128 %0, %2, off\n\t"
            "global_load_async_to_lds_b128 %0, %2, off offset:16\n\t"
            "global_load_async_to_lds_b128 %1, %3, off\n\t"
            "global_load_async_to_lds_b128 %1, %3, off offset:16"
            :: "v"(av), "v"(bv), "v"(ga), "v"(gb) : "memory");
    };

    issue_tile(0, 0);
    asm volatile("s_wait_asynccnt 0x0" ::: "memory");
    __syncthreads();

    const int hs  = lane >> 4;   // half-wave select
    const int l16 = lane & 15;

    for (int kt = 0; kt < nK; ++kt) {
        const int buf = kt & 1;
        if (kt + 1 < nK) issue_tile(kt + 1, buf ^ 1);   // overlap with WMMA below

        union Frag { v16h v; v8h h[2]; };
        Frag af[4], bf[2];
        #pragma unroll
        for (int mi = 0; mi < 4; ++mi) {
            const int m = wm * 64 + mi * 16 + l16;
            af[mi].h[0] = *(const v8h*)&As[buf][m][hs * 8];        // K 0-7 / 8-15
            af[mi].h[1] = *(const v8h*)&As[buf][m][hs * 8 + 16];   // K 16-23 / 24-31
        }
        #pragma unroll
        for (int ni = 0; ni < 2; ++ni) {
            const int n = wn * 32 + ni * 16 + l16;
            bf[ni].h[0] = *(const v8h*)&Bs[buf][n][hs * 8];        // K 0-7 / 8-15
            bf[ni].h[1] = *(const v8h*)&Bs[buf][n][hs * 8 + 16];   // K 16-23 / 24-31
        }
        #pragma unroll
        for (int mi = 0; mi < 4; ++mi)
            #pragma unroll
            for (int ni = 0; ni < 2; ++ni)
                acc[mi][ni] = __builtin_amdgcn_wmma_f32_16x16x32_f16(
                    false, af[mi].v, false, bf[ni].v,
                    (short)0, acc[mi][ni], false, false);

        asm volatile("s_wait_asynccnt 0x0" ::: "memory");
        __syncthreads();
    }

    // ---- epilogue: C/D layout -> global (f32), N guarded here ----
    const int rsel = hs * 8;
    #pragma unroll
    for (int mi = 0; mi < 4; ++mi) {
        #pragma unroll
        for (int ni = 0; ni < 2; ++ni) {
            const int gcn = n0 + wn * 32 + ni * 16 + l16;
            if (gcn >= N) continue;
            const float bvv = bias ? bias[gcn] : 0.0f;
            const int gm = m0 + wm * 64 + mi * 16 + rsel;
            #pragma unroll
            for (int i = 0; i < 8; ++i)
                C[(size_t)(gm + i) * N + gcn] = acc[mi][ni][i] + bvv;
        }
    }
}

// =====================================================================
// elementwise / reduction helpers
// =====================================================================
__global__ void cast_f32_to_f16(const float* __restrict__ in,
                                half_t* __restrict__ out, long long n)
{
    long long i = (long long)blockIdx.x * 256 + threadIdx.x;
    if (i < n) out[i] = (half_t)in[i];
}

// in[R][C] (f32) -> out[C][R] (f16), batched over blockIdx.z
__global__ void cast_transpose(const float* __restrict__ in,
                               half_t* __restrict__ out, int R, int C)
{
    in  += (size_t)blockIdx.z * R * C;
    out += (size_t)blockIdx.z * R * C;
    __shared__ float tile[32][33];
    const int cx = blockIdx.x * 32;   // C
    const int ry = blockIdx.y * 32;   // R
    const int tx = threadIdx.x, ty = threadIdx.y;
    for (int i = ty; i < 32; i += 8) {
        const int r = ry + i, c = cx + tx;
        tile[i][tx] = (r < R && c < C) ? in[(size_t)r * C + c] : 0.0f;
    }
    __syncthreads();
    for (int i = ty; i < 32; i += 8) {
        const int c = cx + i, r = ry + tx;
        if (c < C && r < R) out[(size_t)c * R + r] = (half_t)tile[tx][i];
    }
}

__global__ void embed_kernel(const int* __restrict__ idx,
                             const float* __restrict__ wte,
                             const float* __restrict__ wpe,
                             float* __restrict__ h)
{
    long long i = (long long)blockIdx.x * 256 + threadIdx.x;
    if (i >= (long long)BT_ * E_) return;
    const int e = (int)(i % E_);
    const int row = (int)(i / E_);
    const int t = row % T_;
    h[i] = wte[(size_t)idx[row] * E_ + e] + wpe[(size_t)t * E_ + e];
}

__global__ __launch_bounds__(256)
void ln_cast_kernel(const float* __restrict__ x, const float* __restrict__ g,
                    const float* __restrict__ b, half_t* __restrict__ out)
{
    const int row = blockIdx.x;
    const float* xr = x + (size_t)row * E_;
    __shared__ float red[256];
    float s = 0.f, s2 = 0.f;
    for (int e = threadIdx.x; e < E_; e += 256) {
        const float v = xr[e];
        s += v; s2 += v * v;
    }
    red[threadIdx.x] = s; __syncthreads();
    for (int o = 128; o > 0; o >>= 1) {
        if (threadIdx.x < o) red[threadIdx.x] += red[threadIdx.x + o];
        __syncthreads();
    }
    const float mean = red[0] * (1.0f / E_); __syncthreads();
    red[threadIdx.x] = s2; __syncthreads();
    for (int o = 128; o > 0; o >>= 1) {
        if (threadIdx.x < o) red[threadIdx.x] += red[threadIdx.x + o];
        __syncthreads();
    }
    const float var = red[0] * (1.0f / E_) - mean * mean;
    const float r = rsqrtf(var + 1e-5f);
    for (int e = threadIdx.x; e < E_; e += 256)
        out[(size_t)row * E_ + e] = (half_t)((xr[e] - mean) * r * g[e] + b[e]);
}

// qkv[BT][3E] -> q[bh][t][d], k[bh][t][d], vT[bh][d][t]   (all f16)
__global__ void split_qkv_kernel(const float* __restrict__ qkv,
                                 half_t* __restrict__ qh,
                                 half_t* __restrict__ kh,
                                 half_t* __restrict__ vth)
{
    long long i = (long long)blockIdx.x * 256 + threadIdx.x;
    if (i >= (long long)BT_ * E_) return;
    const int e = (int)(i % E_);
    const int row = (int)(i / E_);
    const int t = row % T_, bb = row / T_;
    const int hh = e / D_, d = e % D_;
    const int bh = bb * H_ + hh;
    const float* base = qkv + (size_t)row * E3_;
    qh [((size_t)bh * T_ + t) * D_ + d] = (half_t)base[e];
    kh [((size_t)bh * T_ + t) * D_ + d] = (half_t)base[E_ + e];
    vth[((size_t)bh * D_ + d) * T_ + t] = (half_t)base[2 * E_ + e];
}

// causal softmax over scores rows; write f16 probs
__global__ __launch_bounds__(256)
void softmax_causal_kernel(const float* __restrict__ S, half_t* __restrict__ P)
{
    const int t  = blockIdx.x;
    const int bh = blockIdx.y;
    const float* sr = S + ((size_t)bh * T_ + t) * T_;
    half_t* pr = P + ((size_t)bh * T_ + t) * T_;
    const float scale = 0.125f;   // D^-0.5
    __shared__ float red[256];
    float mx = -3.4e38f;
    for (int j = threadIdx.x; j <= t; j += 256) mx = fmaxf(mx, sr[j] * scale);
    red[threadIdx.x] = mx; __syncthreads();
    for (int o = 128; o > 0; o >>= 1) {
        if (threadIdx.x < o) red[threadIdx.x] = fmaxf(red[threadIdx.x], red[threadIdx.x + o]);
        __syncthreads();
    }
    const float m = red[0]; __syncthreads();
    float s = 0.f;
    for (int j = threadIdx.x; j <= t; j += 256) s += expf(sr[j] * scale - m);
    red[threadIdx.x] = s; __syncthreads();
    for (int o = 128; o > 0; o >>= 1) {
        if (threadIdx.x < o) red[threadIdx.x] += red[threadIdx.x + o];
        __syncthreads();
    }
    const float inv = 1.0f / red[0];
    for (int j = threadIdx.x; j < T_; j += 256)
        pr[j] = (j <= t) ? (half_t)(expf(sr[j] * scale - m) * inv) : (half_t)0.0f;
}

// attn[bh][t][d] (f32) -> attnh[b*T+t][h*D+d] (f16)
__global__ void merge_heads_kernel(const float* __restrict__ attn,
                                   half_t* __restrict__ attnh)
{
    long long i = (long long)blockIdx.x * 256 + threadIdx.x;
    if (i >= (long long)BT_ * E_) return;
    const int e = (int)(i % E_);
    const int row = (int)(i / E_);
    const int t = row % T_, bb = row / T_;
    const int hh = e / D_, d = e % D_;
    attnh[i] = (half_t)attn[(((size_t)(bb * H_ + hh)) * T_ + t) * D_ + d];
}

__global__ void residual_add_kernel(float* __restrict__ h,
                                    const float* __restrict__ y, long long n)
{
    long long i = (long long)blockIdx.x * 256 + threadIdx.x;
    if (i < n) h[i] += y[i];
}

__global__ void gelu_cast_kernel(const float* __restrict__ u,
                                 half_t* __restrict__ uh, long long n)
{
    long long i = (long long)blockIdx.x * 256 + threadIdx.x;
    if (i < n) {
        const float x = u[i];
        uh[i] = (half_t)(0.5f * x * (1.0f + erff(x * 0.70710678f)));
    }
}

__global__ void zero_kernel(float* __restrict__ p) { p[0] = 0.0f; }

__global__ __launch_bounds__(256)
void loss_kernel(const float* __restrict__ logits, const int* __restrict__ tgt,
                 float* __restrict__ acc)
{
    const int row = blockIdx.x;
    const float* lr = logits + (size_t)row * V_;
    __shared__ float red[256];
    float mx = -3.4e38f;
    for (int j = threadIdx.x; j < V_; j += 256) mx = fmaxf(mx, lr[j]);
    red[threadIdx.x] = mx; __syncthreads();
    for (int o = 128; o > 0; o >>= 1) {
        if (threadIdx.x < o) red[threadIdx.x] = fmaxf(red[threadIdx.x], red[threadIdx.x + o]);
        __syncthreads();
    }
    const float m = red[0]; __syncthreads();
    float s = 0.f;
    for (int j = threadIdx.x; j < V_; j += 256) s += expf(lr[j] - m);
    red[threadIdx.x] = s; __syncthreads();
    for (int o = 128; o > 0; o >>= 1) {
        if (threadIdx.x < o) red[threadIdx.x] += red[threadIdx.x + o];
        __syncthreads();
    }
    if (threadIdx.x == 0)
        atomicAdd(acc, (m + logf(red[0])) - lr[tgt[row]]);
}

__global__ void finalize_loss_kernel(const float* __restrict__ acc,
                                     float* __restrict__ out)
{
    out[0] = acc[0] * (1.0f / (float)BT_);
}

// =====================================================================
// host orchestration
// =====================================================================
static inline int cdiv(long long a, long long b) { return (int)((a + b - 1) / b); }

extern "C" void kernel_launch(void* const* d_in, const int* in_sizes, int n_in,
                              void* d_out, int out_size, void* d_ws, size_t ws_size,
                              hipStream_t stream)
{
    (void)in_sizes; (void)n_in; (void)out_size; (void)ws_size;
    const int*   idx    = (const int*)  d_in[0];
    const int*   tgt    = (const int*)  d_in[1];
    const float* wte    = (const float*)d_in[2];
    const float* wpe    = (const float*)d_in[3];
    const float* ln1_g  = (const float*)d_in[4];
    const float* ln1_b  = (const float*)d_in[5];
    const float* qkv_w  = (const float*)d_in[6];
    const float* qkv_b  = (const float*)d_in[7];
    const float* o_w    = (const float*)d_in[8];
    const float* o_b    = (const float*)d_in[9];
    const float* ln2_g  = (const float*)d_in[10];
    const float* ln2_b  = (const float*)d_in[11];
    const float* up_w   = (const float*)d_in[12];
    const float* up_b   = (const float*)d_in[13];
    const float* down_w = (const float*)d_in[14];
    const float* down_b = (const float*)d_in[15];
    const float* lnf_g  = (const float*)d_in[16];
    const float* lnf_b  = (const float*)d_in[17];
    const float* lm_b   = (const float*)d_in[18];
    float* logits = (float*)d_out;                       // [BT][V]
    float* loss_out = logits + (size_t)BT_ * V_;         // scalar

    // ---- carve workspace ----
    char* p = (char*)d_ws;
    auto carve = [&](size_t bytes) -> void* {
        void* r = (void*)p;
        p += (bytes + 255) & ~(size_t)255;
        return r;
    };
    half_t* wqkvt = (half_t*)carve(sizeof(half_t) * (size_t)L_ * E3_ * E_);
    half_t* wot   = (half_t*)carve(sizeof(half_t) * (size_t)L_ * E_ * E_);
    half_t* wupt  = (half_t*)carve(sizeof(half_t) * (size_t)L_ * E4_ * E_);
    half_t* wdnt  = (half_t*)carve(sizeof(half_t) * (size_t)L_ * E_ * E4_);
    half_t* wteh  = (half_t*)carve(sizeof(half_t) * (size_t)V_ * E_);
    float*  hbuf  = (float*) carve(sizeof(float)  * (size_t)BT_ * E_);
    half_t* xh    = (half_t*)carve(sizeof(half_t) * (size_t)BT_ * E_);
    float*  qkvb  = (float*) carve(sizeof(float)  * (size_t)BT_ * E3_);
    half_t* qh    = (half_t*)carve(sizeof(half_t) * (size_t)B_ * H_ * T_ * D_);
    half_t* kh    = (half_t*)carve(sizeof(half_t) * (size_t)B_ * H_ * T_ * D_);
    half_t* vth   = (half_t*)carve(sizeof(half_t) * (size_t)B_ * H_ * D_ * T_);
    float*  sc    = (float*) carve(sizeof(float)  * (size_t)B_ * H_ * T_ * T_);
    half_t* prb   = (half_t*)carve(sizeof(half_t) * (size_t)B_ * H_ * T_ * T_);
    float*  attn  = (float*) carve(sizeof(float)  * (size_t)B_ * H_ * T_ * D_);
    half_t* attnh = (half_t*)carve(sizeof(half_t) * (size_t)BT_ * E_);
    float*  ybuf  = (float*) carve(sizeof(float)  * (size_t)BT_ * E_);
    float*  ubuf  = (float*) carve(sizeof(float)  * (size_t)BT_ * E4_);
    half_t* uh    = (half_t*)carve(sizeof(half_t) * (size_t)BT_ * E4_);
    float*  acc   = (float*) carve(sizeof(float));

    const dim3 blk256(256);

    // ---- one-time weight prep: cast (+transpose) to fp16 Bt[N][K] ----
    {
        long long n = (long long)V_ * E_;
        cast_f32_to_f16<<<cdiv(n, 256), blk256, 0, stream>>>(wte, wteh, n);
    }
    {
        dim3 b(32, 8);
        cast_transpose<<<dim3(cdiv(E3_,32), cdiv(E_,32),  L_), b, 0, stream>>>(qkv_w,  wqkvt, E_,  E3_);
        cast_transpose<<<dim3(cdiv(E_, 32), cdiv(E_,32),  L_), b, 0, stream>>>(o_w,    wot,   E_,  E_ );
        cast_transpose<<<dim3(cdiv(E4_,32), cdiv(E_,32),  L_), b, 0, stream>>>(up_w,   wupt,  E_,  E4_);
        cast_transpose<<<dim3(cdiv(E_, 32), cdiv(E4_,32), L_), b, 0, stream>>>(down_w, wdnt,  E4_, E_ );
    }

    // ---- embedding ----
    embed_kernel<<<cdiv((long long)BT_ * E_, 256), blk256, 0, stream>>>(idx, wte, wpe, hbuf);

    const long long ne  = (long long)BT_ * E_;
    const long long ne4 = (long long)BT_ * E4_;

    for (int l = 0; l < L_; ++l) {
        // attention
        ln_cast_kernel<<<BT_, blk256, 0, stream>>>(hbuf, ln1_g + (size_t)l * E_, ln1_b + (size_t)l * E_, xh);
        gemm_bt_wmma<<<dim3(cdiv(E3_,128), BT_/128, 1), blk256, 0, stream>>>(
            xh, wqkvt + (size_t)l * E3_ * E_, qkvb, qkv_b + (size_t)l * E3_,
            BT_, E3_, E_, 0, 0, 0, 0);
        split_qkv_kernel<<<cdiv(ne, 256), blk256, 0, stream>>>(qkvb, qh, kh, vth);
        // scores = Q K^T (batched over B*H), fully-masked causal blocks skipped
        gemm_bt_wmma<<<dim3(T_/128, T_/128, B_*H_), blk256, 0, stream>>>(
            qh, kh, sc, nullptr, T_, T_, D_,
            (long long)T_*D_, (long long)T_*D_, (long long)T_*T_, 1);
        softmax_causal_kernel<<<dim3(T_, B_*H_), blk256, 0, stream>>>(sc, prb);
        // attn = P V  (Bt = V^T, prepared in split)
        gemm_bt_wmma<<<dim3(1, T_/128, B_*H_), blk256, 0, stream>>>(
            prb, vth, attn, nullptr, T_, D_, T_,
            (long long)T_*T_, (long long)D_*T_, (long long)T_*D_, 0);
        merge_heads_kernel<<<cdiv(ne, 256), blk256, 0, stream>>>(attn, attnh);
        gemm_bt_wmma<<<dim3(cdiv(E_,128), BT_/128, 1), blk256, 0, stream>>>(
            attnh, wot + (size_t)l * E_ * E_, ybuf, o_b + (size_t)l * E_,
            BT_, E_, E_, 0, 0, 0, 0);
        residual_add_kernel<<<cdiv(ne, 256), blk256, 0, stream>>>(hbuf, ybuf, ne);
        // mlp
        ln_cast_kernel<<<BT_, blk256, 0, stream>>>(hbuf, ln2_g + (size_t)l * E_, ln2_b + (size_t)l * E_, xh);
        gemm_bt_wmma<<<dim3(cdiv(E4_,128), BT_/128, 1), blk256, 0, stream>>>(
            xh, wupt + (size_t)l * E4_ * E_, ubuf, up_b + (size_t)l * E4_,
            BT_, E4_, E_, 0, 0, 0, 0);
        gelu_cast_kernel<<<cdiv(ne4, 256), blk256, 0, stream>>>(ubuf, uh, ne4);
        gemm_bt_wmma<<<dim3(cdiv(E_,128), BT_/128, 1), blk256, 0, stream>>>(
            uh, wdnt + (size_t)l * E_ * E4_, ybuf, down_b + (size_t)l * E_,
            BT_, E_, E4_, 0, 0, 0, 0);
        residual_add_kernel<<<cdiv(ne, 256), blk256, 0, stream>>>(hbuf, ybuf, ne);
    }

    // final LN + tied LM head (wteh is already Bt[V][E])
    ln_cast_kernel<<<BT_, blk256, 0, stream>>>(hbuf, lnf_g, lnf_b, xh);
    gemm_bt_wmma<<<dim3(cdiv(V_,128), BT_/128, 1), blk256, 0, stream>>>(
        xh, wteh, logits, lm_b, BT_, V_, E_, 0, 0, 0, 0);

    // loss = mean NLL
    zero_kernel<<<1, 1, 0, stream>>>(acc);
    loss_kernel<<<BT_, blk256, 0, stream>>>(logits, tgt, acc);
    finalize_loss_kernel<<<1, 1, 0, stream>>>(acc, loss_out);
}
